// Feedback_24137716204037
// MI455X (gfx1250) — matrix operand order repaired and checked by reading, not compiled, not run on hardware
//
#include <hip/hip_runtime.h>
#include <math.h>

// MI455X (gfx1250) implementation.
// B=256, T=128, W=32, H=512, F=1024.
//
// Strategy:
//  - Persistent LSTM kernel: 64 WGs (4 row-groups x 16 h-col slices), 191 steps,
//    device-wide barrier between steps. Wr slice (512x128 bf16 = 128KB) lives in
//    LDS in WMMA-fragment order for the whole kernel. h ping-pongs through L2 as
//    bf16 (256KB) and is staged into LDS each step by the Tensor Data Mover
//    (tensor_load_to_lds + s_wait_tensorcnt) instead of a VALU copy loop.
//  - GEMMs use v_wmma_f32_16x16x32_bf16 (wave32, f32 accumulate).
//  - Candidate branch: relu(inputs@Wc) materialized as bf16 'cat', then a
//    bandwidth-bound WMMA GEMM over the 64MB Wcc (~2.8us at 23.3 TB/s) with
//    double-buffered 2-D TDM tile loads overlapping compute.

typedef __bf16 bf16_t;
typedef __attribute__((ext_vector_type(16))) __bf16 v16bf;
typedef __attribute__((ext_vector_type(8)))  float  v8f;
typedef unsigned int u32x4 __attribute__((ext_vector_type(4)));

// TDM descriptor SGPR groups (this toolchain: clang-23 6-arg builtin:
//   (uint32x4 g0, int32x8 g1, int32x4 g2, int32x4 g3, int32x8 g4, i32 cpol))
typedef unsigned int tdm_u32x4 __attribute__((ext_vector_type(4)));
typedef int          tdm_i32x8 __attribute__((ext_vector_type(8)));
typedef int          tdm_i32x4 __attribute__((ext_vector_type(4)));

#define NWG_LSTM 64
#define NSTEP    191   // 96 warm + 95 decoder cell steps
#define NWARM    96

__device__ __forceinline__ float sigmoidf_(float x) { return 1.0f / (1.0f + __expf(-x)); }

// ---------------------------------------------------------------------------
// Tensor Data Mover: 2-D tile load Global -> LDS (D# per CDNA5 ISA 08, §8).
//   lds_ptr      : generic pointer into LDS (low 32 bits = wave-relative addr)
//   gptr         : global tile start
//   tile_d0/d1   : tile extents in elements (d0 = contiguous dim)
//   tensor_d0/d1 : tensor extents (OOB clamp); here == tile extents
//   stride0      : row stride in elements (48-bit)
//   ds_code      : data_size: 0=1B,1=2B,2=4B,3=8B
// ---------------------------------------------------------------------------
__device__ __forceinline__ void tdm_load_to_lds(void* lds_ptr, const void* gptr,
                                                unsigned tile_d0, unsigned tile_d1,
                                                unsigned tensor_d0, unsigned tensor_d1,
                                                unsigned long long stride0,
                                                int ds_code) {
  unsigned long long ga = (unsigned long long)gptr;
  unsigned lds_off = (unsigned)(unsigned long long)lds_ptr;  // addr[31:0] = LDS offset
  tdm_u32x4 g0;
  g0[0] = 1u;                                                // count=1, user descriptor
  g0[1] = lds_off;                                           // lds_addr [63:32]
  g0[2] = (unsigned)(ga & 0xffffffffu);                      // global_addr [95:64]
  g0[3] = (unsigned)((ga >> 32) & 0x01ffffffu) | (2u << 30); // addr[120:96] | type=2
  tdm_i32x8 g1;
  g1[0] = (ds_code & 3) << 16;                               // wg_mask=0, data_size
  g1[1] = (int)((tensor_d0 & 0xffffu) << 16);                // tensor_dim0[15:0] @63:48
  g1[2] = (int)(((tensor_d0 >> 16) & 0xffffu) |              // tensor_dim0[31:16]
                ((tensor_d1 & 0xffffu) << 16));              // tensor_dim1[15:0]
  g1[3] = (int)(((tensor_d1 >> 16) & 0xffffu) |              // tensor_dim1[31:16]
                ((tile_d0 & 0xffffu) << 16));                // tile_dim0 @127:112
  g1[4] = (int)(tile_d1 & 0xffffu);                          // tile_dim1; tile_dim2=0
  g1[5] = (int)(unsigned)(stride0 & 0xffffffffu);            // dim0_stride[31:0]
  g1[6] = (int)(unsigned)((stride0 >> 32) & 0xffffu);        // dim0_stride[47:32]
  g1[7] = 0;
  tdm_i32x4 z4 = {0, 0, 0, 0};                               // groups 2/3 unused (2-D)
  tdm_i32x8 z8 = {0, 0, 0, 0, 0, 0, 0, 0};                   // second-desc group (unused)
  __builtin_amdgcn_tensor_load_to_lds(g0, g1, z4, z4, z8, 0);
}

__device__ __forceinline__ void grid_barrier(unsigned* cnt, int phase) {
  // release: make this WG's h stores visible at device scope (waits storecnt, wb)
  __builtin_amdgcn_fence(__ATOMIC_RELEASE, "agent");
  __syncthreads();
  if (threadIdx.x == 0) {
    __hip_atomic_fetch_add(&cnt[phase], 1u, __ATOMIC_RELAXED, __HIP_MEMORY_SCOPE_AGENT);
    while (__hip_atomic_load(&cnt[phase], __ATOMIC_RELAXED, __HIP_MEMORY_SCOPE_AGENT) <
           (unsigned)NWG_LSTM) {
      __builtin_amdgcn_s_sleep(1);
    }
  }
  __syncthreads();
  // acquire: invalidate WGP$ so stale h lines are not re-read
  __builtin_amdgcn_fence(__ATOMIC_ACQUIRE, "agent");
}

// ---------------------------------------------------------------------------
// Persistent LSTM kernel. grid=64 WGs x 256 threads (8 waves).
// WG tile: 64 rows x 32 h-cols (x 4 gates). Wave tile: 16 rows x 16 cols,
// holding i/f/g/o accumulators + persistent c tile.
// Dynamic LDS layout (bytes):
//   [0,131072)        wr_lds  : Wr slice, bf16, WMMA-B fragment order
//   [131072,196608)   h_lds   : 64x512 bf16 A tile (current h rows)
//   [196608,198144)   wkb     : 128 x {Wk0,Wk1,bias} f32
//   [198144,202240)   ww_lds  : Ww 512x2 f32
//   [202240,202752)   x_lds   : 64x2 f32 current input
// ---------------------------------------------------------------------------
__global__ __launch_bounds__(256, 1)
void lstm_kernel(const float* __restrict__ inp, const float* __restrict__ Wk,
                 const float* __restrict__ Wr, const float* __restrict__ bias,
                 const float* __restrict__ Ww, const float* __restrict__ bw,
                 bf16_t* __restrict__ h0, bf16_t* __restrict__ h1,
                 bf16_t* __restrict__ catb, float* __restrict__ preds,
                 unsigned* __restrict__ cnt) {
  extern __shared__ char smem[];
  bf16_t* wr_lds = (bf16_t*)smem;
  bf16_t* h_lds  = (bf16_t*)(smem + 131072);
  float*  wkb    = (float*)(smem + 196608);
  float*  ww_lds = (float*)(smem + 198144);
  float*  x_lds  = (float*)(smem + 202240);

  const int tid  = threadIdx.x;
  const int lane = tid & 31;
  const int wave = tid >> 5;
  const int rsub = wave & 3;   // 16-row subtile
  const int hseg = wave >> 2;  // 16-col subtile of the 32-col slice
  const int rowWG = blockIdx.x & 3;
  const int colWG = blockIdx.x >> 2;
  const int rowBase = rowWG * 64;
  const int colBase = colWG * 32;

  // ---- one-time: Wr slice -> LDS in B-fragment order -----------------------
  // frag linear index = (((kb*4+g)*2+hseg)*32 + lane)*16 + e
  // B layout (32x16 bf16): col = lane&15, K = (lane>=16?16:0)+e
  for (int idx = tid; idx < 65536; idx += 256) {
    int e  = idx & 15;
    int ln = (idx >> 4) & 31;
    int hs = (idx >> 9) & 1;
    int g  = (idx >> 10) & 3;
    int kb = idx >> 12;
    int K  = kb * 32 + ((ln >= 16) ? 16 : 0) + e;
    int col = g * 512 + colBase + hs * 16 + (ln & 15);
    wr_lds[idx] = (bf16_t)Wr[K * 2048 + col];
  }
  if (tid < 128) {  // Wk + bias for our 128 gate-columns
    int g = tid >> 5, jj = tid & 31;
    int col = g * 512 + colBase + jj;
    wkb[tid * 3 + 0] = Wk[col];
    wkb[tid * 3 + 1] = Wk[2048 + col];
    wkb[tid * 3 + 2] = bias[col];
  }
  for (int idx = tid; idx < 1024; idx += 256) ww_lds[idx] = Ww[idx];
  __syncthreads();

  v8f creg = {};  // persistent cell state tile (f32)

  for (int step = 0; step < NSTEP; ++step) {
    bf16_t* hsrc = (step & 1) ? h1 : h0;
    bf16_t* hdst = (step & 1) ? h0 : h1;

    // stage h rows [64][512] bf16 -> LDS via TDM (single 64KB contiguous tile)
    if (wave == 0) {
      tdm_load_to_lds(h_lds, hsrc + rowBase * 512,
                      /*tile*/ 16384, 1, /*tensor*/ 16384, 1,
                      /*stride0*/ 16384ull, /*4B*/ 2);
      __builtin_amdgcn_s_wait_tensorcnt(0);
    }
    __syncthreads();

    // current input x (warm: from inputs; decoder: pred recomputed from h_lds)
    if (step < NWARM) {
      if (tid < 128) {
        int r = tid >> 1, comp = tid & 1;
        x_lds[tid] = inp[(rowBase + r) * 256 + (32 + step) * 2 + comp];
      }
    } else {
      if (tid < 128) {
        int r = tid >> 1, comp = tid & 1;
        const bf16_t* hr = h_lds + r * 512;
        float s = 0.f;
        for (int k = 0; k < 512; ++k) s += (float)hr[k] * ww_lds[k * 2 + comp];
        s += bw[comp];
        x_lds[tid] = s;
        if (colWG == 0)  // unique writer of predictions[step-96]
          preds[(rowBase + r) * 192 + (step - NWARM) * 2 + comp] = s;
      }
      if (step == NWARM && colWG == 0) {
        // warm_out -> first 512 columns of 'cat' (bf16), straight from LDS
        for (int idx = tid; idx < 32768; idx += 256) {
          int r = idx >> 9, k = idx & 511;
          catb[(rowBase + r) * 16384 + k] = h_lds[idx];
        }
      }
    }
    __syncthreads();

    // ---- GEMM: 4 gate tiles, K=512 (16 x K32 WMMA steps) -------------------
    v8f acc[4] = {};
    const int arow = (lane & 15) + rsub * 16;
    const bf16_t* hrow = h_lds + arow * 512;
    const int aoff = (lane >= 16) ? 16 : 0;  // A layout lane-half K offset
#pragma unroll 4
    for (int kb = 0; kb < 16; ++kb) {
      v16bf afrag;
      int kbase = kb * 32 + aoff;
#pragma unroll
      for (int e = 0; e < 16; ++e) afrag[e] = hrow[kbase + e];
#pragma unroll
      for (int g = 0; g < 4; ++g) {
        union { u32x4 u[2]; v16bf v; } bfu;
        const u32x4* bp =
            (const u32x4*)(wr_lds + ((((kb * 4 + g) * 2 + hseg) * 32) + lane) * 16);
        bfu.u[0] = bp[0];
        bfu.u[1] = bp[1];
        acc[g] = __builtin_amdgcn_wmma_f32_16x16x32_bf16(
            false, afrag, false, bfu.v, (short)0, acc[g], false, false);
      }
    }

    // ---- epilogue: + bias + x@Wk, LSTM cell, store h (bf16) ----------------
    {
      const int colL = hseg * 16 + (lane & 15);
      const int roff = ((lane >= 16) ? 8 : 0) + rsub * 16;
#pragma unroll
      for (int ve = 0; ve < 8; ++ve) {
        int rl = ve + roff;
        float x0 = x_lds[rl * 2 + 0];
        float x1 = x_lds[rl * 2 + 1];
        const float* wi = wkb + (colL) * 3;
        const float* wf = wkb + (32 + colL) * 3;
        const float* wg = wkb + (64 + colL) * 3;
        const float* wo = wkb + (96 + colL) * 3;
        float zi = acc[0][ve] + wi[0] * x0 + wi[1] * x1 + wi[2];
        float zf = acc[1][ve] + wf[0] * x0 + wf[1] * x1 + wf[2];
        float zg = acc[2][ve] + wg[0] * x0 + wg[1] * x1 + wg[2];
        float zo = acc[3][ve] + wo[0] * x0 + wo[1] * x1 + wo[2];
        float cn = sigmoidf_(zf) * creg[ve] + sigmoidf_(zi) * tanhf(zg);
        float hn = sigmoidf_(zo) * tanhf(cn);
        creg[ve] = cn;
        hdst[(rowBase + rl) * 512 + colBase + colL] = (bf16_t)hn;
      }
    }
    grid_barrier(cnt, step);
  }

  // final prediction (index 95) from h after step 190 (lives in h1)
  if (colWG == 0) {
    if (wave == 0) {
      tdm_load_to_lds(h_lds, h1 + rowBase * 512, 16384, 1, 16384, 1, 16384ull, 2);
      __builtin_amdgcn_s_wait_tensorcnt(0);
    }
    __syncthreads();
    if (tid < 128) {
      int r = tid >> 1, comp = tid & 1;
      const bf16_t* hr = h_lds + r * 512;
      float s = 0.f;
      for (int k = 0; k < 512; ++k) s += (float)hr[k] * ww_lds[k * 2 + comp];
      preds[(rowBase + r) * 192 + 95 * 2 + comp] = s + bw[comp];
    }
  }
}

// ---------------------------------------------------------------------------
// cat columns 512.. : relu(inputs[:, :31] @ Wc + bc) flattened, as bf16.
// grid = dim3(62, 256) : y = batch row, x*256+tid = flat column (31*512=15872)
// ---------------------------------------------------------------------------
__global__ __launch_bounds__(256)
void prep_cat_kernel(const float* __restrict__ inp, const float* __restrict__ Wc,
                     const float* __restrict__ bc, bf16_t* __restrict__ catb) {
  int r = blockIdx.y;
  int cIdx = blockIdx.x * 256 + threadIdx.x;  // 0..15871
  int t = cIdx >> 9;
  int j = cIdx & 511;
  float x0 = inp[r * 256 + t * 2 + 0];
  float x1 = inp[r * 256 + t * 2 + 1];
  float v = fmaxf(x0 * Wc[j] + x1 * Wc[512 + j] + bc[j], 0.f);
  catb[r * 16384 + 512 + cIdx] = (bf16_t)v;
}

// ---------------------------------------------------------------------------
// ct = relu(cat @ Wcc + bcc). M=256,N=1024,K=16384. Bandwidth-bound on Wcc
// (64MB fp32 -> ~2.8us floor at 23.3 TB/s). Double-buffered 2-D TDM tile
// loads (A: 64x64B rows, W: 32x256B rows) overlap DMA with WMMA compute;
// s_wait_tensorcnt 2 retires only the older tile pair.
// grid = 64 WGs (4 row x 16 col), WG tile 64x64, wave: 16x16 x 2 col-tiles.
// ---------------------------------------------------------------------------
__global__ __launch_bounds__(256, 1)
void cat_gemm_kernel(const bf16_t* __restrict__ catb, const float* __restrict__ Wcc,
                     const float* __restrict__ bcc, float* __restrict__ ct) {
  __shared__ bf16_t a_lds[2][64 * 32];  // 8KB : A tiles (bf16)
  __shared__ float  w_lds[2][32 * 64];  // 16KB: Wcc tiles (f32, converted on use)
  const int tid = threadIdx.x;
  const int lane = tid & 31;
  const int wave = tid >> 5;
  const int rsub = wave & 3;
  const int cpair = wave >> 2;
  const int rowBase = (blockIdx.x & 3) * 64;
  const int colBase = (blockIdx.x >> 2) * 64;

  v8f acc0 = {}, acc1 = {};
  const int arow = (lane & 15) + rsub * 16;
  const int aoff = (lane >= 16) ? 16 : 0;
  const int koff = (lane >= 16) ? 16 : 0;
  const int col0 = (cpair * 2 + 0) * 16 + (lane & 15);
  const int col1 = (cpair * 2 + 1) * 16 + (lane & 15);

  if (wave == 0) {  // preload kb=0 tile pair
    tdm_load_to_lds(&a_lds[0][0], catb + rowBase * 16384,
                    32, 64, 32, 64, 16384ull, /*2B*/ 1);
    tdm_load_to_lds(&w_lds[0][0], Wcc + colBase,
                    64, 32, 64, 32, 1024ull, /*4B*/ 2);
  }

  for (int kb = 0; kb < 512; ++kb) {
    const int cb = kb & 1;
    __syncthreads();  // everyone done reading buffer cb^1 from previous iter
    if (wave == 0) {
      if (kb + 1 < 512) {
        tdm_load_to_lds(&a_lds[cb ^ 1][0],
                        catb + rowBase * 16384 + (kb + 1) * 32,
                        32, 64, 32, 64, 16384ull, 1);
        tdm_load_to_lds(&w_lds[cb ^ 1][0],
                        Wcc + (kb + 1) * 32 * 1024 + colBase,
                        64, 32, 64, 32, 1024ull, 2);
        __builtin_amdgcn_s_wait_tensorcnt(2);  // older pair (buffer cb) complete
      } else {
        __builtin_amdgcn_s_wait_tensorcnt(0);
      }
    }
    __syncthreads();  // buffer cb ready for all waves

    v16bf afrag, b0, b1;
    const bf16_t* ar = &a_lds[cb][0] + arow * 32;
#pragma unroll
    for (int e = 0; e < 16; ++e) afrag[e] = ar[aoff + e];
    const float* wp = &w_lds[cb][0];
#pragma unroll
    for (int e = 0; e < 16; ++e) {
      b0[e] = (bf16_t)wp[(koff + e) * 64 + col0];
      b1[e] = (bf16_t)wp[(koff + e) * 64 + col1];
    }
    acc0 = __builtin_amdgcn_wmma_f32_16x16x32_bf16(false, afrag, false, b0, (short)0,
                                                   acc0, false, false);
    acc1 = __builtin_amdgcn_wmma_f32_16x16x32_bf16(false, afrag, false, b1, (short)0,
                                                   acc1, false, false);
  }

  const int roff = ((lane >= 16) ? 8 : 0) + rsub * 16;
#pragma unroll
  for (int ve = 0; ve < 8; ++ve) {
    int r = rowBase + roff + ve;
    ct[r * 1024 + colBase + col0] = fmaxf(acc0[ve] + bcc[colBase + col0], 0.f);
    ct[r * 1024 + colBase + col1] = fmaxf(acc1[ve] + bcc[colBase + col1], 0.f);
  }
}

// candidate_prob = sigmoid(ct@Wp+bp), candidate_cost = ct@Wco+bco
__global__ __launch_bounds__(256)
void finish_kernel(const float* __restrict__ ct, const float* __restrict__ Wp,
                   const float* __restrict__ bp, const float* __restrict__ Wco,
                   const float* __restrict__ bco, float* __restrict__ cost_out,
                   float* __restrict__ prob_out) {
  int r = threadIdx.x;
  float s0 = 0.f, s1 = 0.f, s2 = 0.f;
  const float* row = ct + r * 1024;
  for (int k = 0; k < 1024; ++k) {
    float v = row[k];
    s0 += v * Wp[k * 2];
    s1 += v * Wp[k * 2 + 1];
    s2 += v * Wco[k];
  }
  cost_out[r] = s2 + bco[0];
  prob_out[r * 2 + 0] = sigmoidf_(s0 + bp[0]);
  prob_out[r * 2 + 1] = sigmoidf_(s1 + bp[1]);
}

__global__ __launch_bounds__(256)
void init_kernel(unsigned* __restrict__ h0, unsigned* __restrict__ cnt) {
  int idx = blockIdx.x * 256 + threadIdx.x;  // 65536 threads
  h0[idx] = 0u;                              // zero h_0 (256KB)
  if (idx < 256) cnt[idx] = 0u;              // zero barrier counters
}

__global__ __launch_bounds__(256)
void copyin_kernel(const float* __restrict__ inp, float* __restrict__ out) {
  int idx = blockIdx.x * 256 + threadIdx.x;  // 65536 elements
  out[idx] = inp[idx];
}

// ---------------------------------------------------------------------------
extern "C" void kernel_launch(void* const* d_in, const int* in_sizes, int n_in,
                              void* d_out, int out_size, void* d_ws, size_t ws_size,
                              hipStream_t stream) {
  const float* inputs = (const float*)d_in[0];
  const float* Wk  = (const float*)d_in[1];
  const float* Wr  = (const float*)d_in[2];
  const float* b   = (const float*)d_in[3];
  const float* Ww  = (const float*)d_in[4];
  const float* bw  = (const float*)d_in[5];
  const float* Wc  = (const float*)d_in[6];
  const float* bc  = (const float*)d_in[7];
  const float* Wcc = (const float*)d_in[8];
  const float* bcc = (const float*)d_in[9];
  const float* Wp  = (const float*)d_in[10];
  const float* bp  = (const float*)d_in[11];
  const float* Wco = (const float*)d_in[12];
  const float* bco = (const float*)d_in[13];
  float* out = (float*)d_out;

  // workspace layout (needs ~9.6 MB)
  char* ws = (char*)d_ws;
  bf16_t*   h0   = (bf16_t*)(ws + 0);         // 256KB
  bf16_t*   h1   = (bf16_t*)(ws + 262144);    // 256KB
  bf16_t*   catb = (bf16_t*)(ws + 524288);    // 256x16384 bf16 = 8MB
  float*    ct   = (float*)(ws + 8912896);    // 256x1024 f32 = 1MB
  unsigned* cnt  = (unsigned*)(ws + 9961472); // 1KB barrier counters

  init_kernel<<<256, 256, 0, stream>>>((unsigned*)h0, cnt);
  lstm_kernel<<<NWG_LSTM, 256, 202752, stream>>>(inputs, Wk, Wr, b, Ww, bw, h0, h1,
                                                 catb, out /*predictions*/, cnt);
  prep_cat_kernel<<<dim3(62, 256), 256, 0, stream>>>(inputs, Wc, bc, catb);
  cat_gemm_kernel<<<64, 256, 0, stream>>>(catb, Wcc, bcc, ct);
  finish_kernel<<<1, 256, 0, stream>>>(ct, Wp, bp, Wco, bco,
                                       out + 49152 /*cost*/, out + 49408 /*prob*/);
  copyin_kernel<<<256, 256, 0, stream>>>(inputs, out + 49920 /*inputs passthrough*/);
}
